// BCTransformer_73263552135320
// MI455X (gfx1250) — matrix operand order repaired and checked by reading, not compiled, not run on hardware
//
#include <hip/hip_runtime.h>
#include <hip/hip_bf16.h>
#include <math.h>

typedef __attribute__((ext_vector_type(16))) _Float16 v16h;
typedef __attribute__((ext_vector_type(8)))  _Float16 v8h;
typedef __attribute__((ext_vector_type(8)))  float    v8f;

#define BD 16          // batch
#define TT 512         // seq len
#define DD 768         // model dim
#define NHH 12         // heads
#define DHH 64         // head dim
#define LL 6           // layers
#define MTOK (BD*TT)   // 8192 tokens

__device__ __forceinline__ v8f wmma_f16(v16h a, v16h b, v8f c) {
  return __builtin_amdgcn_wmma_f32_16x16x32_f16(false, a, false, b, (short)0, c, false, false);
}

__device__ __forceinline__ v16h cat8(v8h lo, v8h hi) {
  return __builtin_shufflevector(lo, hi, 0,1,2,3,4,5,6,7,8,9,10,11,12,13,14,15);
}

// ---------------- prologue kernels ----------------

__global__ void deck_kernel(const int* __restrict__ deck, const int* __restrict__ opp,
                            const float* __restrict__ tok, float* __restrict__ comb) {
  int idx = blockIdx.x * blockDim.x + threadIdx.x;
  if (idx >= BD * DD) return;
  int b = idx / DD, d = idx % DD;
  float s = 0.f;
  for (int i = 0; i < 8; i++)
    s += tok[(size_t)deck[b*8+i]*DD + d] + tok[(size_t)opp[b*8+i]*DD + d];
  comb[idx] = s * 0.125f;   // mean(deck) + mean(opp) = (sum_d + sum_o)/8
}

__global__ void mask_kernel(const int* __restrict__ hc, float* __restrict__ maskadd,
                            int* __restrict__ lastidx) {
  int b = threadIdx.x;
  if (b >= BD) return;
  bool any = false;
  for (int t = 0; t < TT; t++) any |= (hc[b*TT + t] != 0);
  int cnt = 0;
  for (int t = 0; t < TT; t++) {
    bool v = (hc[b*TT + t] != 0);
    if (t == 0 && !any) v = true;   // all-pad fixup
    maskadd[b*TT + t] = v ? 0.f : -1e30f;
    cnt += v ? 1 : 0;
  }
  int li = cnt - 1; if (li < 0) li = 0;
  lastidx[b] = li;
}

__global__ void embed_kernel(const int* __restrict__ hc, const int* __restrict__ hp,
                             const float* __restrict__ tok, const float* __restrict__ pemb,
                             const float* __restrict__ pos, const float* __restrict__ comb,
                             float* __restrict__ x) {
  size_t idx = (size_t)blockIdx.x * blockDim.x + threadIdx.x;
  if (idx >= (size_t)MTOK * DD) return;
  int d = idx % DD;
  size_t bt = idx / DD;
  int t = bt % TT, b = bt / TT;
  int card = hc[bt];
  int pl = hp[bt]; pl = pl < 0 ? 0 : (pl > 1 ? 1 : pl);
  // x_emb[PAD_X] and y_emb[PAD_Y] rows are zero -> omitted
  x[idx] = tok[(size_t)card*DD + d] + pemb[pl*DD + d] + comb[b*DD + d] + pos[(size_t)t*DD + d];
}

// f32 W[K][N] -> f16 WT[N][K]
__global__ void convtrans_kernel(const float* __restrict__ in, _Float16* __restrict__ out,
                                 int K, int N) {
  size_t idx = (size_t)blockIdx.x * blockDim.x + threadIdx.x;
  if (idx >= (size_t)K * N) return;
  int n = idx / K, k = idx % K;
  out[idx] = (_Float16)in[(size_t)k * N + n];
}

// ---------------- layernorm (wave per row) ----------------

__global__ void ln_kernel(const float* __restrict__ x, const float* __restrict__ s,
                          const float* __restrict__ bp, _Float16* __restrict__ out, int rows) {
  int row  = blockIdx.x * (blockDim.x >> 5) + (threadIdx.x >> 5);
  int lane = threadIdx.x & 31;
  if (row >= rows) return;
  const float* xr = x + (size_t)row * DD;
  float sm = 0.f, sq = 0.f;
  for (int i = lane; i < DD; i += 32) { float v = xr[i]; sm += v; sq += v*v; }
  for (int off = 1; off < 32; off <<= 1) {
    sm += __shfl_xor(sm, off, 32);
    sq += __shfl_xor(sq, off, 32);
  }
  float mean = sm * (1.f/DD);
  float var  = sq * (1.f/DD) - mean*mean;
  float rstd = rsqrtf(var + 1e-5f);
  _Float16* orow = out + (size_t)row * DD;
  for (int i = lane; i < DD; i += 32)
    orow[i] = (_Float16)((xr[i] - mean) * rstd * s[i] + bp[i]);
}

// ---------------- WMMA GEMM: C[M,N] = A[M,K](f16) @ WT[N,K]^T + bias ----------------
// FLAGS: 1 = store f16, 2 = exact GELU, 4 = residual accumulate into f32 outf
// Software-pipelined: K-tile k+1 fragments are loaded while tile k's WMMAs issue.

template<int FLAGS>
__global__ void gemm_wmma_kernel(const _Float16* __restrict__ A,
                                 const _Float16* __restrict__ WT,
                                 const float* __restrict__ bias,
                                 _Float16* __restrict__ out16,
                                 float* __restrict__ outf,
                                 int M, int N, int K) {
  const int lane = threadIdx.x & 31;
  const int wave = threadIdx.x >> 5;
  const int hf = lane >> 4, lr = lane & 15;
  const int m0 = blockIdx.y * 128 + wave * 32;
  const int n0 = blockIdx.x * 64;
  if (m0 >= M) return;

  v8f c[2][4] = {};

  // per-lane base pointers (k offset added per tile)
  const _Float16* ap0 = A + (size_t)(m0 + lr) * K + hf*16;
  const _Float16* ap1 = A + (size_t)(m0 + 16 + lr) * K + hf*16;
  const _Float16* bp0 = WT + (size_t)(n0 +  0 + lr) * K + hf*16;
  const _Float16* bp1 = WT + (size_t)(n0 + 16 + lr) * K + hf*16;
  const _Float16* bp2 = WT + (size_t)(n0 + 32 + lr) * K + hf*16;
  const _Float16* bp3 = WT + (size_t)(n0 + 48 + lr) * K + hf*16;

  v16h a_cur[2], b_cur[4], a_nxt[2], b_nxt[4];
  a_cur[0] = *(const v16h*)ap0;
  a_cur[1] = *(const v16h*)ap1;
  b_cur[0] = *(const v16h*)bp0;
  b_cur[1] = *(const v16h*)bp1;
  b_cur[2] = *(const v16h*)bp2;
  b_cur[3] = *(const v16h*)bp3;

  for (int k0 = 0; k0 < K; k0 += 32) {
    const int k1 = k0 + 32;
    if (k1 < K) {
      // issue next tile's loads before consuming current tile
      a_nxt[0] = *(const v16h*)(ap0 + k1);
      a_nxt[1] = *(const v16h*)(ap1 + k1);
      b_nxt[0] = *(const v16h*)(bp0 + k1);
      b_nxt[1] = *(const v16h*)(bp1 + k1);
      b_nxt[2] = *(const v16h*)(bp2 + k1);
      b_nxt[3] = *(const v16h*)(bp3 + k1);
      if (k1 + 32 < K) {
        __builtin_prefetch(ap0 + k1 + 32, 0, 3);
        __builtin_prefetch(bp0 + k1 + 32, 0, 3);
        __builtin_prefetch(bp2 + k1 + 32, 0, 3);
      }
    }
#pragma unroll
    for (int mi = 0; mi < 2; mi++)
#pragma unroll
      for (int ni = 0; ni < 4; ni++)
        c[mi][ni] = wmma_f16(a_cur[mi], b_cur[ni], c[mi][ni]);
#pragma unroll
    for (int i = 0; i < 2; i++) a_cur[i] = a_nxt[i];
#pragma unroll
    for (int i = 0; i < 4; i++) b_cur[i] = b_nxt[i];
  }

#pragma unroll
  for (int mi = 0; mi < 2; mi++)
#pragma unroll
    for (int ni = 0; ni < 4; ni++) {
      int col = n0 + ni*16 + lr;
      float bv = bias[col];
#pragma unroll
      for (int r = 0; r < 8; r++) {
        int row = m0 + mi*16 + r + 8*hf;
        float v = c[mi][ni][r] + bv;
        if (FLAGS & 2) v = 0.5f * v * (1.f + erff(v * 0.70710678118654752f));
        if (FLAGS & 4)      outf[(size_t)row * N + col] += v;
        else if (FLAGS & 1) out16[(size_t)row * N + col] = (_Float16)v;
        else                outf[(size_t)row * N + col] = v;
      }
    }
}

// ---------------- V transpose: qkv[b,t, 1536 + h*64+d] -> vt[(b,h,d), t] ----------------

__global__ void vtrans_kernel(const _Float16* __restrict__ qkv, _Float16* __restrict__ vt) {
  size_t idx = (size_t)blockIdx.x * blockDim.x + threadIdx.x;
  if (idx >= (size_t)BD * NHH * DHH * TT) return;
  int t = idx & (TT-1);
  size_t r = idx >> 9;
  int d = r & (DHH-1); r >>= 6;
  int h = r % NHH;
  int b = r / NHH;
  vt[idx] = qkv[((size_t)(b*TT) + t) * (3*DD) + 2*DD + h*DHH + d];
}

// ---------------- flash attention, one wave per (b, h, 16-query tile) ----------------
// (round-1 structure: fragments loaded at point of use -> no register spills)

__global__ void attn_kernel(const _Float16* __restrict__ qkv,   // [MTOK][3D]
                            const _Float16* __restrict__ vt,    // [(b,h,d)][T]
                            const float* __restrict__ maskadd,  // [B][T]
                            _Float16* __restrict__ yo) {        // [MTOK][D]
  const int lane = threadIdx.x & 31;
  const int hf = lane >> 4, lr = lane & 15;
  const int q0 = blockIdx.x * 16;
  const int h  = blockIdx.y;
  const int b  = blockIdx.z;
  const float scale = 0.125f;   // 1/sqrt(64)

  __shared__ __align__(32) _Float16 Pt[16][32];

  // Q fragments (16 queries x 64 dims -> two K=32 chunks)
  v16h aq[2];
  {
    const _Float16* qp = qkv + ((size_t)(b*TT) + q0 + lr) * (3*DD) + h*DHH;
#pragma unroll
    for (int cc = 0; cc < 2; cc++) {
      const _Float16* p = qp + cc*32 + hf*16;
      aq[cc] = *(const v16h*)p;
    }
  }

  float mrow[8], lrow[8], alpha[8];
#pragma unroll
  for (int r = 0; r < 8; r++) { mrow[r] = -1e30f; lrow[r] = 0.f; }
  v8f o[4] = {};

  const int nchunk = (q0 + 15) / 32 + 1;
  for (int kc = 0; kc < nchunk; kc++) {
    const int kbase = kc * 32;
    float p0[8], p1[8];
#pragma unroll
    for (int j = 0; j < 2; j++) {
      const int ks = kbase + j*16;
      v8f sfr = {};
#pragma unroll
      for (int cc = 0; cc < 2; cc++) {
        v16h bk = *(const v16h*)(qkv + ((size_t)(b*TT) + ks + lr) * (3*DD)
                                 + DD + h*DHH + cc*32 + hf*16);
        sfr = wmma_f16(aq[cc], bk, sfr);
      }
      const int key = ks + lr;
      float madd = maskadd[b*TT + key];
      float* pj = j ? p1 : p0;
#pragma unroll
      for (int r = 0; r < 8; r++) {
        int qrow = q0 + r + 8*hf;
        float sv = sfr[r] * scale + madd;
        if (key > qrow) sv = -1e30f;   // causal
        pj[r] = sv;
      }
    }
    // online softmax update (row reductions across 16 lanes of same half)
#pragma unroll
    for (int r = 0; r < 8; r++) {
      float rm = fmaxf(p0[r], p1[r]);
      for (int off = 1; off < 16; off <<= 1) rm = fmaxf(rm, __shfl_xor(rm, off, 32));
      float nm = fmaxf(mrow[r], rm);
      float al = __expf(mrow[r] - nm);
      float e0 = __expf(p0[r] - nm);
      float e1 = __expf(p1[r] - nm);
      float rs = e0 + e1;
      for (int off = 1; off < 16; off <<= 1) rs += __shfl_xor(rs, off, 32);
      lrow[r] = lrow[r] * al + rs;
      mrow[r] = nm;
      alpha[r] = al;
      p0[r] = e0; p1[r] = e1;
    }
#pragma unroll
    for (int nj = 0; nj < 4; nj++)
#pragma unroll
      for (int r = 0; r < 8; r++)
        o[nj][r] *= alpha[r];
    // P (C-layout) -> LDS -> A-layout
#pragma unroll
    for (int r = 0; r < 8; r++) {
      Pt[r + 8*hf][lr]      = (_Float16)p0[r];
      Pt[r + 8*hf][16 + lr] = (_Float16)p1[r];
    }
    v16h ap = *(const v16h*)&Pt[lr][hf*16];
#pragma unroll
    for (int nj = 0; nj < 4; nj++) {
      v16h bv = *(const v16h*)(vt + ((size_t)((b*NHH + h)*DHH + nj*16 + lr)) * TT
                               + kbase + 16*hf);
      o[nj] = wmma_f16(ap, bv, o[nj]);
    }
  }

#pragma unroll
  for (int r = 0; r < 8; r++) {
    float inv = lrow[r] > 0.f ? 1.f / lrow[r] : 0.f;
    int row = q0 + r + 8*hf;
#pragma unroll
    for (int nj = 0; nj < 4; nj++)
      yo[((size_t)(b*TT) + row) * DD + h*DHH + nj*16 + lr] = (_Float16)(o[nj][r] * inv);
  }
}

// ---------------- final: gather last token, LN, 768x9 head ----------------

__global__ void head_kernel(const float* __restrict__ x, const int* __restrict__ lastidx,
                            const float* __restrict__ lnfs, const float* __restrict__ lnfb,
                            const float* __restrict__ hw, const float* __restrict__ hb,
                            float* __restrict__ out) {
  int b = blockIdx.x, tid = threadIdx.x;
  __shared__ float row[DD];
  __shared__ float red[256];
  const float* xr = x + ((size_t)(b*TT) + lastidx[b]) * DD;
  float s = 0.f, sq = 0.f;
  for (int i = tid; i < DD; i += 256) { float v = xr[i]; s += v; sq += v*v; }
  red[tid] = s; __syncthreads();
  for (int st = 128; st > 0; st >>= 1) { if (tid < st) red[tid] += red[tid + st]; __syncthreads(); }
  float mean = red[0] / DD; __syncthreads();
  red[tid] = sq; __syncthreads();
  for (int st = 128; st > 0; st >>= 1) { if (tid < st) red[tid] += red[tid + st]; __syncthreads(); }
  float var = red[0] / DD - mean*mean;
  float rstd = rsqrtf(var + 1e-5f);
  __syncthreads();
  for (int i = tid; i < DD; i += 256)
    row[i] = (xr[i] - mean) * rstd * lnfs[i] + lnfb[i];
  __syncthreads();
  if (tid < 9) {
    float acc = hb[tid];
    for (int d = 0; d < DD; d++) acc += row[d] * hw[d*9 + tid];
    out[b*9 + tid] = acc;
  }
}

// ---------------- host orchestration ----------------

extern "C" void kernel_launch(void* const* d_in, const int* in_sizes, int n_in,
                              void* d_out, int out_size, void* d_ws, size_t ws_size,
                              hipStream_t stream) {
  (void)in_sizes; (void)n_in; (void)out_size; (void)ws_size;
  const int*   hc    = (const int*)d_in[0];
  const int*   hp    = (const int*)d_in[1];
  const int*   deck  = (const int*)d_in[2];
  const int*   opp   = (const int*)d_in[3];
  const float* tok   = (const float*)d_in[4];
  const float* pemb  = (const float*)d_in[5];
  const float* pos   = (const float*)d_in[6];
  const float* ln1s  = (const float*)d_in[9];
  const float* ln1b  = (const float*)d_in[10];
  const float* qkvw  = (const float*)d_in[11];
  const float* qkvb  = (const float*)d_in[12];
  const float* projw = (const float*)d_in[13];
  const float* projb = (const float*)d_in[14];
  const float* ln2s  = (const float*)d_in[15];
  const float* ln2b  = (const float*)d_in[16];
  const float* fc1w  = (const float*)d_in[17];
  const float* fc1b  = (const float*)d_in[18];
  const float* fc2w  = (const float*)d_in[19];
  const float* fc2b  = (const float*)d_in[20];
  const float* lnfs  = (const float*)d_in[21];
  const float* lnfb  = (const float*)d_in[22];
  const float* hw    = (const float*)d_in[23];
  const float* hb    = (const float*)d_in[24];

  char* ws = (char*)d_ws;
  size_t off = 0;
  auto alloc = [&](size_t bytes) -> char* {
    char* p = ws + off;
    off = (off + bytes + 255) & ~(size_t)255;
    return p;
  };

  const size_t QKVW = (size_t)DD*3*DD, PROJW = (size_t)DD*DD,
               FC1W = (size_t)DD*4*DD, FC2W  = (size_t)4*DD*DD;
  const size_t LSTRIDE = QKVW + PROJW + FC1W + FC2W;   // f16 elems per layer

  _Float16* wT    = (_Float16*)alloc(LSTRIDE * LL * sizeof(_Float16));
  float*    x     = (float*)   alloc((size_t)MTOK * DD * sizeof(float));
  _Float16* xn16  = (_Float16*)alloc((size_t)MTOK * DD * sizeof(_Float16));
  _Float16* qkv16 = (_Float16*)alloc((size_t)MTOK * 3*DD * sizeof(_Float16));
  _Float16* vt16  = (_Float16*)alloc((size_t)BD*NHH*DHH*TT * sizeof(_Float16));
  _Float16* y16   = (_Float16*)alloc((size_t)MTOK * DD * sizeof(_Float16));
  _Float16* hmid16 = qkv16;   // alias: qkv16+vt16 region (50.3MB) reused for fc1 output
  float*    comb    = (float*)alloc((size_t)BD * DD * sizeof(float));
  float*    maskadd = (float*)alloc((size_t)BD * TT * sizeof(float));
  int*      lastidx = (int*)  alloc(64);
  float*    outp    = (float*)d_out;

  // prologue
  deck_kernel<<<(BD*DD + 255)/256, 256, 0, stream>>>(deck, opp, tok, comb);
  mask_kernel<<<1, 32, 0, stream>>>(hc, maskadd, lastidx);
  embed_kernel<<<((size_t)MTOK*DD + 255)/256, 256, 0, stream>>>(hc, hp, tok, pemb, pos, comb, x);

  // convert + transpose weights to f16 [N][K]
  for (int l = 0; l < LL; l++) {
    _Float16* base = wT + (size_t)l * LSTRIDE;
    convtrans_kernel<<<(QKVW + 255)/256, 256, 0, stream>>>(qkvw + (size_t)l*QKVW, base, DD, 3*DD);
    convtrans_kernel<<<(PROJW + 255)/256, 256, 0, stream>>>(projw + (size_t)l*PROJW, base + QKVW, DD, DD);
    convtrans_kernel<<<(FC1W + 255)/256, 256, 0, stream>>>(fc1w + (size_t)l*FC1W, base + QKVW + PROJW, DD, 4*DD);
    convtrans_kernel<<<(FC2W + 255)/256, 256, 0, stream>>>(fc2w + (size_t)l*FC2W, base + QKVW + PROJW + FC1W, 4*DD, DD);
  }

  const int LN_BLOCKS = MTOK / 8;  // 8 waves per 256-thread block, wave per row
  for (int l = 0; l < LL; l++) {
    _Float16* qkvT = wT + (size_t)l * LSTRIDE;
    _Float16* projT = qkvT + QKVW;
    _Float16* fc1T  = projT + PROJW;
    _Float16* fc2T  = fc1T + FC1W;

    // attention block
    ln_kernel<<<LN_BLOCKS, 256, 0, stream>>>(x, ln1s + l*DD, ln1b + l*DD, xn16, MTOK);
    {
      dim3 g(3*DD/64, MTOK/128);
      gemm_wmma_kernel<1><<<g, 128, 0, stream>>>(xn16, qkvT, qkvb + l*3*DD, qkv16, nullptr, MTOK, 3*DD, DD);
    }
    vtrans_kernel<<<((size_t)BD*NHH*DHH*TT + 255)/256, 256, 0, stream>>>(qkv16, vt16);
    {
      dim3 g(TT/16, NHH, BD);
      attn_kernel<<<g, 32, 0, stream>>>(qkv16, vt16, maskadd, y16);
    }
    {
      dim3 g(DD/64, MTOK/128);
      gemm_wmma_kernel<4><<<g, 128, 0, stream>>>(y16, projT, projb + l*DD, nullptr, x, MTOK, DD, DD);
    }
    // MLP block
    ln_kernel<<<LN_BLOCKS, 256, 0, stream>>>(x, ln2s + l*DD, ln2b + l*DD, xn16, MTOK);
    {
      dim3 g(4*DD/64, MTOK/128);
      gemm_wmma_kernel<3><<<g, 128, 0, stream>>>(xn16, fc1T, fc1b + l*4*DD, hmid16, nullptr, MTOK, 4*DD, DD);
    }
    {
      dim3 g(DD/64, MTOK/128);
      gemm_wmma_kernel<4><<<g, 128, 0, stream>>>(hmid16, fc2T, fc2b + l*DD, nullptr, x, MTOK, DD, 4*DD);
    }
  }

  head_kernel<<<BD, 256, 0, stream>>>(x, lastidx, lnfs, lnfb, hw, hb, outp);
}
